// GNN_8761733284301
// MI455X (gfx1250) — compile-verified, matrix operand order checked by reading, not run on hardware
//
#include <hip/hip_runtime.h>
#include <hip/hip_bf16.h>

typedef __attribute__((ext_vector_type(16))) __bf16         v16bf;
typedef __attribute__((ext_vector_type(8)))  float          v8f;
typedef __attribute__((ext_vector_type(4)))  float          v4f;
typedef __attribute__((ext_vector_type(8)))  unsigned short v8u;
typedef __attribute__((ext_vector_type(16))) unsigned short v16u;
typedef __attribute__((ext_vector_type(4)))  unsigned short v4u;

__device__ __forceinline__ unsigned short f2bf(float f) {
    // native f32 -> bf16 convert (RNE); lowers to v_cvt_pk_bf16_f32
    return __builtin_bit_cast(unsigned short, (__bf16)f);
}

__device__ __forceinline__ void atomic_fadd(float* p, float v) {
    __hip_atomic_fetch_add(p, v, __ATOMIC_RELAXED, __HIP_MEMORY_SCOPE_AGENT);
}

// ---------------------------------------------------------------------------
// Zero accumulators, init degree to 1.0 (self loop).
// ---------------------------------------------------------------------------
__global__ __launch_bounds__(256) void k_init(float* __restrict__ agg1,
                                              float* __restrict__ agg2,
                                              float* __restrict__ deg, int N) {
    int t = blockIdx.x * blockDim.x + threadIdx.x;
    v4f z = {0.f, 0.f, 0.f, 0.f};
    if (t < N * 16) ((v4f*)agg1)[t] = z;      // N*64 floats
    if (t < N * 4)  ((v4f*)agg2)[t] = z;      // N*16 floats
    if (t < N)      deg[t] = 1.0f;            // self-loop
}

// ---------------------------------------------------------------------------
// In-degree count over edges (dst row of edge_index).
// ---------------------------------------------------------------------------
__global__ __launch_bounds__(256) void k_deg(const long long* __restrict__ ei,
                                             float* __restrict__ deg, int E) {
    int t = blockIdx.x * blockDim.x + threadIdx.x;
    if (t < E) {
        long long d = ei[(long long)E + t];
        atomic_fadd(&deg[d], 1.0f);
    }
}

__global__ __launch_bounds__(256) void k_dinv(float* __restrict__ deg, int N) {
    int t = blockIdx.x * blockDim.x + threadIdx.x;
    if (t < N) {
        float d = deg[t];                     // >= 1 always
        deg[t] = __frsqrt_rn(d);
    }
}

// ---------------------------------------------------------------------------
// Tiled bf16 WMMA GEMM: Y[Nrows x NC] = X[Nrows x K] * W[K x NC]
// Block = 256 threads = 8 waves; block owns 128 rows, each wave 16 rows.
// W transposed in LDS as bf16 [NC][K]; X double-buffered as bf16 [2][128][32].
// Fragment layout per CDNA5 16-bit A/B tables: lane half h, element i ->
// K = i + 8*(h + (i>>3)): two 8-element (16B) runs at 8*h and 16+8*h.
// X loads are non-temporal (single-use stream; keep L2 for graph data).
// Full-tile test is block-uniform so 781/782 blocks take the unguarded path.
// ---------------------------------------------------------------------------
template <int K, int NC>
__global__ __launch_bounds__(256) void k_gemm_bf16(const float* __restrict__ X,
                                                   const float* __restrict__ Wg,
                                                   float* __restrict__ Y,
                                                   int Nrows) {
    constexpr int NT     = NC / 16;
    constexpr int NSTEPS = K / 32;
    __shared__ __align__(16) unsigned short sW[NC * K];
    __shared__ __align__(16) unsigned short sX[2][128 * 32];

    const int tid  = threadIdx.x;
    const int lane = tid & 31;
    const int wv   = tid >> 5;
    const int r    = lane & 15;     // row (A) / col (B,C)
    const int half = lane >> 4;
    const int tileBase = blockIdx.x * 128;
    const bool full = (tileBase + 128 <= Nrows);   // block-uniform

    // Stage W transposed: global [K][NC] -> LDS [NC][K], bf16
    for (int i = tid; i < NC * K; i += 256) {
        int k = i / NC, n = i % NC;
        sW[n * K + k] = f2bf(Wg[i]);
    }

    // this thread's 4 staging slots: float4 index idx4 = tid + v*256
    int srow[4], scol[4];
    const float* xbase[4];
    bool inb[4];
    #pragma unroll
    for (int v = 0; v < 4; ++v) {
        int idx4 = tid + v * 256;
        srow[v]  = idx4 >> 3;           // 8 float4 per 32-wide row
        scol[v]  = (idx4 & 7) * 4;
        xbase[v] = X + (long long)(tileBase + srow[v]) * K + scol[v];
        inb[v]   = (tileBase + srow[v] < Nrows);
    }

    v4f reg[4];
    auto load_step = [&](int k0) {
        if (full) {                      // uniform branch: no exec masking
            #pragma unroll
            for (int v = 0; v < 4; ++v)
                reg[v] = __builtin_nontemporal_load((const v4f*)(xbase[v] + k0));
        } else {
            #pragma unroll
            for (int v = 0; v < 4; ++v) {
                if (inb[v]) reg[v] = __builtin_nontemporal_load((const v4f*)(xbase[v] + k0));
                else        reg[v] = (v4f){0.f, 0.f, 0.f, 0.f};
            }
        }
    };

    load_step(0);                        // prologue: k-step 0

    v8f acc[NT] = {};
    int b = 0;

    #pragma unroll
    for (int s = 0; s < NSTEPS; ++s) {
        // commit current registers to LDS buffer b
        #pragma unroll
        for (int v = 0; v < 4; ++v) {
            v4u bv;
            bv[0] = f2bf(reg[v][0]); bv[1] = f2bf(reg[v][1]);
            bv[2] = f2bf(reg[v][2]); bv[3] = f2bf(reg[v][3]);
            *(v4u*)(&sX[b][srow[v] * 32 + scol[v]]) = bv;
        }
        __syncthreads();

        // prefetch next k-step while this one computes
        if (s + 1 < NSTEPS) load_step((s + 1) * 32);

        // A fragment: row = wave's 16-row slice + r; two 8-elem K runs
        const int arow = wv * 16 + r;
        v8u a_lo = *(const v8u*)(&sX[b][arow * 32 + 8 * half]);
        v8u a_hi = *(const v8u*)(&sX[b][arow * 32 + 16 + 8 * half]);
        v16u au;
        #pragma unroll
        for (int i = 0; i < 8; ++i) { au[i] = a_lo[i]; au[i + 8] = a_hi[i]; }
        v16bf A = __builtin_bit_cast(v16bf, au);

        const int k0 = s * 32;
        #pragma unroll
        for (int nt = 0; nt < NT; ++nt) {
            int ncol = nt * 16 + r;
            v8u b_lo = *(const v8u*)(&sW[ncol * K + k0 + 8 * half]);
            v8u b_hi = *(const v8u*)(&sW[ncol * K + k0 + 16 + 8 * half]);
            v16u bu;
            #pragma unroll
            for (int i = 0; i < 8; ++i) { bu[i] = b_lo[i]; bu[i + 8] = b_hi[i]; }
            v16bf Bm = __builtin_bit_cast(v16bf, bu);
            acc[nt] = __builtin_amdgcn_wmma_f32_16x16x32_bf16(
                false, A, false, Bm, (short)0, acc[nt], false, false);
        }
        b ^= 1;
    }

    // C/D layout: VGPR j -> M = j + 8*half, N = r
    if (full) {
        #pragma unroll
        for (int nt = 0; nt < NT; ++nt)
            #pragma unroll
            for (int j = 0; j < 8; ++j) {
                int grow = tileBase + wv * 16 + j + 8 * half;
                Y[(long long)grow * NC + nt * 16 + r] = acc[nt][j];
            }
    } else {
        #pragma unroll
        for (int nt = 0; nt < NT; ++nt)
            #pragma unroll
            for (int j = 0; j < 8; ++j) {
                int grow = tileBase + wv * 16 + j + 8 * half;
                if (grow < Nrows)
                    Y[(long long)grow * NC + nt * 16 + r] = acc[nt][j];
            }
    }
}

// ---------------------------------------------------------------------------
// Edge aggregation, layer 1 (64 features): one wave32 per edge, 2 feat/lane.
// h1 (25.6MB) and agg1 live in L2 (192MB) -> gathers/atomics at L2 rate.
// ---------------------------------------------------------------------------
__global__ __launch_bounds__(256) void k_agg1(const long long* __restrict__ ei,
                                              const float* __restrict__ dinv,
                                              const float* __restrict__ h,
                                              float* __restrict__ agg, int E) {
    int wid  = (blockIdx.x * blockDim.x + threadIdx.x) >> 5;
    int lane = threadIdx.x & 31;
    if (wid >= E) return;
    long long s = ei[wid];
    long long d = ei[(long long)E + wid];
    float norm = dinv[s] * dinv[d];
    float2 v = *(const float2*)(h + s * 64 + lane * 2);
    float* p = agg + d * 64 + lane * 2;
    atomic_fadd(p,     v.x * norm);
    atomic_fadd(p + 1, v.y * norm);
}

// agg1 += self-loop + bias, then ReLU (in place -> becomes GEMM2 input)
__global__ __launch_bounds__(256) void k_fin1(const float* __restrict__ h1,
                                              const float* __restrict__ dinv,
                                              const float* __restrict__ b1,
                                              float* __restrict__ agg1, int N) {
    int t = blockIdx.x * blockDim.x + threadIdx.x;
    if (t >= N * 64) return;
    int n = t >> 6, f = t & 63;
    float di = dinv[n];
    float v = agg1[t] + h1[t] * di * di + b1[f];
    agg1[t] = v > 0.f ? v : 0.f;
}

// ---------------------------------------------------------------------------
// Edge aggregation, layer 2 (16 features): one thread per (edge, feature).
// ---------------------------------------------------------------------------
__global__ __launch_bounds__(256) void k_agg2(const long long* __restrict__ ei,
                                              const float* __restrict__ dinv,
                                              const float* __restrict__ h2,
                                              float* __restrict__ agg2, int E) {
    int t = blockIdx.x * blockDim.x + threadIdx.x;
    if (t >= E * 16) return;
    int e = t >> 4, f = t & 15;
    long long s = ei[e];
    long long d = ei[(long long)E + e];
    float norm = dinv[s] * dinv[d];
    atomic_fadd(&agg2[d * 16 + f], h2[s * 16 + f] * norm);
}

// self-loop + bias + log_softmax over 16 cols; one thread per node
__global__ __launch_bounds__(256) void k_fin2(const float* __restrict__ h2,
                                              const float* __restrict__ dinv,
                                              const float* __restrict__ b2,
                                              const float* __restrict__ agg2,
                                              float* __restrict__ out, int N) {
    int n = blockIdx.x * blockDim.x + threadIdx.x;
    if (n >= N) return;
    float di = dinv[n];
    float dd = di * di;
    float v[16];
    float m = -INFINITY;
    #pragma unroll
    for (int f = 0; f < 16; ++f) {
        float x = agg2[n * 16 + f] + h2[n * 16 + f] * dd + b2[f];
        v[f] = x;
        m = fmaxf(m, x);
    }
    float s = 0.f;
    #pragma unroll
    for (int f = 0; f < 16; ++f) s += __expf(v[f] - m);
    float ls = __logf(s);
    #pragma unroll
    for (int f = 0; f < 16; ++f) out[n * 16 + f] = v[f] - m - ls;
}

// ---------------------------------------------------------------------------
extern "C" void kernel_launch(void* const* d_in, const int* in_sizes, int n_in,
                              void* d_out, int out_size, void* d_ws, size_t ws_size,
                              hipStream_t stream) {
    const float*     x  = (const float*)d_in[0];
    const long long* ei = (const long long*)d_in[1];   // int64 edge_index [2][E]
    const float*     W1 = (const float*)d_in[2];       // [256][64]
    const float*     b1 = (const float*)d_in[3];       // [64]
    const float*     W2 = (const float*)d_in[4];       // [64][16]
    const float*     b2 = (const float*)d_in[5];       // [16]
    float* out = (float*)d_out;

    const int N = in_sizes[0] / 256;
    const int E = in_sizes[1] / 2;

    float* ws   = (float*)d_ws;
    float* h1   = ws;                       // N*64
    float* agg1 = h1   + (size_t)N * 64;    // N*64 (becomes relu'd layer-1 out)
    float* h2   = agg1 + (size_t)N * 64;    // N*16
    float* agg2 = h2   + (size_t)N * 16;    // N*16
    float* dinv = agg2 + (size_t)N * 16;    // N (deg, then 1/sqrt(deg) in place)

    const int B = 256;
    k_init<<<(N * 16 + B - 1) / B, B, 0, stream>>>(agg1, agg2, dinv, N);
    k_deg<<<(E + B - 1) / B, B, 0, stream>>>(ei, dinv, E);
    k_dinv<<<(N + B - 1) / B, B, 0, stream>>>(dinv, N);

    k_gemm_bf16<256, 64><<<(N + 127) / 128, 256, 0, stream>>>(x, W1, h1, N);
    k_agg1<<<(E + 7) / 8, 256, 0, stream>>>(ei, dinv, h1, agg1, E);
    k_fin1<<<(N * 64 + B - 1) / B, B, 0, stream>>>(h1, dinv, b1, agg1, N);

    k_gemm_bf16<64, 16><<<(N + 127) / 128, 256, 0, stream>>>(agg1, W2, h2, N);
    k_agg2<<<(E * 16 + B - 1) / B, B, 0, stream>>>(ei, dinv, h2, agg2, E);
    k_fin2<<<(N + B - 1) / B, B, 0, stream>>>(h2, dinv, b2, agg2, out, N);
}
